// PSWE_54949811585070
// MI455X (gfx1250) — matrix operand order repaired and checked by reading, not compiled, not run on hardware
//
#include <hip/hip_runtime.h>

// Problem constants (match reference)
#define BB   64
#define NSET 4096
#define DD   128
#define MM   1024
#define LL   128

typedef __attribute__((ext_vector_type(16))) __bf16 v16bf;
typedef __attribute__((ext_vector_type(8)))  float  v8f;
typedef __attribute__((ext_vector_type(4)))  unsigned int u32x4;
typedef __attribute__((ext_vector_type(8)))  int   i32x8;
typedef __attribute__((ext_vector_type(4)))  int   i32x4;

// ---------------------------------------------------------------------------
// Kernel 1: weight-norm + bf16 hi/lo split of W.  W[l,d] = v[l,d]/||v[l,:]||
// ---------------------------------------------------------------------------
__global__ __launch_bounds__(DD) void pswe_normalize_split(
    const float* __restrict__ theta, __bf16* __restrict__ Whi, __bf16* __restrict__ Wlo) {
  const int l = blockIdx.x, d = threadIdx.x;
  __shared__ float red[DD];
  float v = theta[l * DD + d];
  red[d] = v * v;
  __syncthreads();
  for (int st = DD / 2; st > 0; st >>= 1) {
    if (d < st) red[d] += red[d + st];
    __syncthreads();
  }
  float w = v / sqrtf(red[0]);
  __bf16 h = (__bf16)w;
  Whi[l * DD + d] = h;
  Wlo[l * DD + d] = (__bf16)(w - (float)h);
}

// ---------------------------------------------------------------------------
// Kernel 2: per-projection constants. For each l:
//   argsort ref_points[:,l]  ->  scatter weight into ws2
//   build alpha[l,n] = sum over interp taps (constant j[m], t[m])
//   c[l] = sum_m ref[m,l]*weight[l,m]
// ---------------------------------------------------------------------------
__global__ __launch_bounds__(256) void pswe_prep_alpha(
    const float* __restrict__ ref, const float* __restrict__ weight,
    float* __restrict__ alpha, float* __restrict__ cvec) {
  const int l = blockIdx.x, tid = threadIdx.x;
  __shared__ float key[MM];
  __shared__ int   idx[MM];
  __shared__ float ws2[MM];
  __shared__ float alpha_s[NSET];
  __shared__ float red[256];

  for (int m = tid; m < MM; m += 256) { key[m] = ref[m * LL + l]; idx[m] = m; }
  __syncthreads();
  // bitonic argsort ascending over M=1024
  for (int k = 2; k <= MM; k <<= 1) {
    for (int j = k >> 1; j > 0; j >>= 1) {
      for (int i = tid; i < MM; i += 256) {
        int ixj = i ^ j;
        if (ixj > i) {
          bool up = ((i & k) == 0);
          float a = key[i], b = key[ixj];
          if ((a > b) == up) {
            key[i] = b; key[ixj] = a;
            int t = idx[i]; idx[i] = idx[ixj]; idx[ixj] = t;
          }
        }
      }
      __syncthreads();
    }
  }
  // ws2[Rind[m,l]] = weight[l,m]
  for (int m = tid; m < MM; m += 256) ws2[idx[m]] = weight[l * MM + m];
  for (int n = tid; n < NSET; n += 256) alpha_s[n] = 0.f;
  __syncthreads();
  // constant interp taps: u = (m+1)*(N+1)/(M+1); j = floor(u)-1; t = frac
  for (int m = tid; m < MM; m += 256) {
    double u = (double)(m + 1) * (double)(NSET + 1) / (double)(MM + 1);
    int j = (int)u - 1;
    float t = (float)(u - (double)(j + 1));
    float w = ws2[m];
    atomicAdd(&alpha_s[j],     (1.f - t) * w);
    atomicAdd(&alpha_s[j + 1], t * w);
  }
  __syncthreads();
  for (int n = tid; n < NSET; n += 256) alpha[(size_t)l * NSET + n] = alpha_s[n];
  // c[l]
  float acc = 0.f;
  for (int m = tid; m < MM; m += 256) acc += ref[m * LL + l] * weight[l * MM + m];
  red[tid] = acc;
  __syncthreads();
  for (int st = 128; st > 0; st >>= 1) {
    if (tid < st) red[tid] += red[tid + st];
    __syncthreads();
  }
  if (tid == 0) cvec[l] = red[0];
}

// ---------------------------------------------------------------------------
// Kernel 3: slicesT[b,l,n] = sum_d X[b,n,d]*W[l,d] via bf16 WMMA, split-precision
// (hi*hi + hi*lo + lo*hi, f32 accumulation -> ~f32 accuracy at bf16 matrix rate).
// Three independent accumulator chains so consecutive WMMAs never have a
// back-to-back RAW on the same D registers (see ISA 7.12.1 hazard table).
// Block = 256 threads = 8 independent waves; each wave owns one 16(n)x16(l) tile.
// ---------------------------------------------------------------------------
__global__ __launch_bounds__(256) void pswe_gemm_slices(
    const float* __restrict__ X, const __bf16* __restrict__ Whi,
    const __bf16* __restrict__ Wlo, float* __restrict__ slicesT) {
  const int lane = threadIdx.x & 31;
  const int wave = threadIdx.x >> 5;
  const int n0 = blockIdx.x * 128 + wave * 16;   // set-dim tile base (GEMM "M")
  const int l0 = blockIdx.y * 16;                // projection tile base (GEMM "N")
  const int b  = blockIdx.z;
  const int half = lane >> 4;                    // lane group (ISA layout)
  const int mr   = lane & 15;

  const float* xrow = X + ((size_t)b * NSET + n0 + mr) * DD;
  v8f accA = {};   // hi*hi chain
  v8f accB = {};   // hi*lo chain
  v8f accC = {};   // lo*hi chain

#pragma unroll
  for (int k0 = 0; k0 < DD; k0 += 32) {
    // --- A tile (16x32): lane<16 -> K {0..15}, lane>=16 -> K {16..31}
    float af[16];
#pragma unroll
    for (int e = 0; e < 8; ++e) af[e]     = xrow[k0 + half * 16 + e];
#pragma unroll
    for (int e = 0; e < 8; ++e) af[8 + e] = xrow[k0 + half * 16 + 8 + e];
    v16bf ahi, alo;
#pragma unroll
    for (int e = 0; e < 16; ++e) {
      __bf16 h = (__bf16)af[e];
      ahi[e] = h;
      alo[e] = (__bf16)(af[e] - (float)h);
    }
    // --- B tile (32x16) = W^T: col = lane%16, K = 16*half + e (contiguous in W row)
    const size_t wbase = (size_t)(l0 + mr) * DD + k0 + half * 16;
    v16bf bhi, blo;
#pragma unroll
    for (int e = 0; e < 16; ++e) { bhi[e] = Whi[wbase + e]; blo[e] = Wlo[wbase + e]; }

    accA = __builtin_amdgcn_wmma_f32_16x16x32_bf16(false, ahi, false, bhi, (short)0, accA, false, false);
    accB = __builtin_amdgcn_wmma_f32_16x16x32_bf16(false, ahi, false, blo, (short)0, accB, false, false);
    accC = __builtin_amdgcn_wmma_f32_16x16x32_bf16(false, alo, false, bhi, (short)0, accC, false, false);
  }
  // C layout: element r -> row (r + 8*half) of the n-tile, col = lane%16 of the l-tile.
  // Transposed store [B,L,N]: per lane 8 contiguous floats along n.
  float* dst = slicesT + ((size_t)b * LL + l0 + mr) * NSET + n0 + half * 8;
#pragma unroll
  for (int r = 0; r < 8; ++r) dst[r] = accA[r] + (accB[r] + accC[r]);
}

// ---------------------------------------------------------------------------
// Kernel 4: per-(b,l) bitonic sort of 4096 floats in LDS, fused weighted
// reduction against alpha[l,:], out[b,l] = c[l] - dot.
// The 16 KB contiguous slice row is DMA'd into LDS by the Tensor Data Mover
// (one descriptor; tracked by TENSORcnt).
// ---------------------------------------------------------------------------
__global__ __launch_bounds__(256) void pswe_sort_reduce(
    const float* __restrict__ slicesT, const float* __restrict__ alpha,
    const float* __restrict__ cvec, float* __restrict__ out) {
  const int blk = blockIdx.x;           // blk = b*LL + l
  const int l = blk % LL;
  const int tid = threadIdx.x;
  const float* src = slicesT + (size_t)blk * NSET;
  const float* al  = alpha + (size_t)l * NSET;
  __shared__ float s[NSET];
  __shared__ float red[256];

#if defined(__gfx1250__)
  if (tid < 32) {  // wave 0 issues the TDM descriptor (TDM ignores EXEC; wave-level op)
    unsigned long long ga = (unsigned long long)(uintptr_t)src;
    unsigned int ldsa = (unsigned int)(uintptr_t)&s[0];  // flat->LDS: low 32 bits
    // D# group 0: count=1 | lds_addr | global_addr[56:0] | type=2
    u32x4 g0 = { 1u, ldsa, (unsigned int)ga,
                 (unsigned int)((ga >> 32) & 0x01FFFFFFu) | (2u << 30) };
    // D# group 1: data_size=4B; tensor_dim0=tile_dim0=NSET; tensor_dim1=tile_dim1=1
    i32x8 g1 = { (int)(2u << 16),                 // data_size=2 (4 bytes)
                 (int)((unsigned)NSET << 16),     // tensor_dim0[15:0]
                 (int)(1u << 16),                 // tensor_dim0 hi=0 | tensor_dim1 lo=1
                 (int)((unsigned)NSET << 16),     // tensor_dim1 hi=0 | tile_dim0=NSET
                 1,                               // tile_dim1=1 | tile_dim2=0
                 NSET,                            // tensor_dim0_stride[31:0]
                 0, 0 };
    i32x4 gz = { 0, 0, 0, 0 };
#if __clang_major__ >= 23
    i32x8 gz8 = { 0, 0, 0, 0, 0, 0, 0, 0 };
    __builtin_amdgcn_tensor_load_to_lds(g0, g1, gz, gz, gz8, 0);
#else
    __builtin_amdgcn_tensor_load_to_lds(g0, g1, gz, gz, 0);
#endif
    __builtin_amdgcn_s_wait_tensorcnt(0);
  }
  // warm alpha row into cache while the DMA completes
  __builtin_prefetch(al + tid * 16, 0, 3);
  __syncthreads();
#else
  for (int i = tid; i < NSET; i += 256) s[i] = src[i];
  __builtin_prefetch(al + tid * 16, 0, 3);
  __syncthreads();
#endif

  for (int k = 2; k <= NSET; k <<= 1) {
    for (int j = k >> 1; j > 0; j >>= 1) {
      for (int i = tid; i < NSET; i += 256) {
        int ixj = i ^ j;
        if (ixj > i) {
          bool up = ((i & k) == 0);
          float a = s[i], b = s[ixj];
          if ((a > b) == up) { s[i] = b; s[ixj] = a; }
        }
      }
      __syncthreads();
    }
  }
  float acc = 0.f;
  for (int i = tid; i < NSET; i += 256) acc += s[i] * al[i];
  red[tid] = acc;
  __syncthreads();
  for (int st = 128; st > 0; st >>= 1) {
    if (tid < st) red[tid] += red[tid + st];
    __syncthreads();
  }
  if (tid == 0) out[blk] = cvec[l] - red[0];
}

// ---------------------------------------------------------------------------
extern "C" void kernel_launch(void* const* d_in, const int* in_sizes, int n_in,
                              void* d_out, int out_size, void* d_ws, size_t ws_size,
                              hipStream_t stream) {
  const float* X      = (const float*)d_in[0];   // [B,N,D]
  const float* theta  = (const float*)d_in[1];   // [L,D]
  const float* ref    = (const float*)d_in[2];   // [M,L]
  const float* weight = (const float*)d_in[3];   // [L,M]
  float* out = (float*)d_out;                    // [B,L]

  char* ws = (char*)d_ws;
  size_t off = 0;
  auto carve = [&](size_t bytes) -> void* {
    void* p = ws + off;
    off += (bytes + 255) & ~(size_t)255;
    return p;
  };
  __bf16* Whi    = (__bf16*)carve((size_t)LL * DD * sizeof(__bf16));
  __bf16* Wlo    = (__bf16*)carve((size_t)LL * DD * sizeof(__bf16));
  float*  alpha  = (float*) carve((size_t)LL * NSET * sizeof(float));
  float*  cvec   = (float*) carve((size_t)LL * sizeof(float));
  float*  slices = (float*) carve((size_t)BB * LL * NSET * sizeof(float));
  (void)ws_size; (void)in_sizes; (void)n_in; (void)out_size;

  pswe_normalize_split<<<LL, DD, 0, stream>>>(theta, Whi, Wlo);
  pswe_prep_alpha<<<LL, 256, 0, stream>>>(ref, weight, alpha, cvec);
  pswe_gemm_slices<<<dim3(NSET / 128, LL / 16, BB), 256, 0, stream>>>(X, Whi, Wlo, slices);
  pswe_sort_reduce<<<BB * LL, 256, 0, stream>>>(slices, alpha, cvec, out);
}
